// CMTBlock_52707838656836
// MI455X (gfx1250) — compile-verified, matrix operand order checked by reading, not run on hardware
//
#include <hip/hip_runtime.h>

// ---------------------------------------------------------------------------
// CMT block for MI455X (gfx1250): wave32 + v_wmma_f32_16x16x32_f16 everywhere
// ---------------------------------------------------------------------------

typedef __attribute__((ext_vector_type(16))) _Float16 v16h;
typedef __attribute__((ext_vector_type(8)))  float    v8f;

#define WMMA_F16(a, b, c) \
  __builtin_amdgcn_wmma_f32_16x16x32_f16(false, (a), false, (b), (short)0, (c), false, false)

static constexpr int kB    = 64;
static constexpr int kC    = 128;
static constexpr int kH    = 28;
static constexpr int kW    = 28;
static constexpr int kHW   = 784;      // 49 * 16 exactly -> no partial q tiles
static constexpr int kHeads= 4;
static constexpr int kD    = 32;       // head dim == one WMMA K step
static constexpr int kE    = 512;
static constexpr int kKVS  = 15;
static constexpr int kL    = 225;      // K/V tokens
static constexpr int kLp   = 240;      // padded to 15 n-tiles of 16
static constexpr int kLp2  = 256;      // padded K for P@V (8 k-steps of 32)
static constexpr int kM    = kB * kHW; // 50176 GEMM rows

__device__ __forceinline__ float gelu_f(float x) {
  return 0.5f * x * (1.0f + erff(x * 0.70710678118654752f));
}

// ---------------------------------------------------------------------------
// small utility kernels
// ---------------------------------------------------------------------------
__global__ void cvt_f32_to_f16(const float* __restrict__ s, _Float16* __restrict__ d, int n) {
  int i = blockIdx.x * 256 + threadIdx.x;
  if (i < n) d[i] = (_Float16)s[i];
}

__global__ void fill_zero_f16(_Float16* __restrict__ d, int n) {
  int i = blockIdx.x * 256 + threadIdx.x;
  if (i < n) d[i] = (_Float16)0.0f;
}

// ---------------------------------------------------------------------------
// LPU: x_lpu = x + dwconv3x3(x).  Read NCHW (contiguous plane), write NHWC.
// one block per (b, c) plane
// ---------------------------------------------------------------------------
__global__ __launch_bounds__(256)
void lpu_kernel(const float* __restrict__ x, const float* __restrict__ w,
                const float* __restrict__ bias, float* __restrict__ xlpu) {
  const int b = blockIdx.x >> 7, c = blockIdx.x & 127;
  const float* p = x + ((size_t)b * kC + c) * kHW;
  float k[9];
#pragma unroll
  for (int i = 0; i < 9; ++i) k[i] = w[c * 9 + i];
  const float bb = bias[c];
  for (int s = threadIdx.x; s < kHW; s += 256) {
    int yy = s / kW, xx = s - yy * kW;
    float a = bb;
#pragma unroll
    for (int ky = 0; ky < 3; ++ky) {
      int iy = yy + ky - 1;
      if (iy < 0 || iy >= kH) continue;
#pragma unroll
      for (int kx = 0; kx < 3; ++kx) {
        int ix = xx + kx - 1;
        if (ix < 0 || ix >= kW) continue;
        a += k[ky * 3 + kx] * p[iy * kW + ix];
      }
    }
    xlpu[((size_t)b * kHW + s) * kC + c] = p[s] + a;
  }
}

// ---------------------------------------------------------------------------
// LayerNorm over (H,W) per (b,c), spatial affine g/b [784]; NHWC f32 -> f16.
// one block per batch b; threads: c = t&127, half = t>>7 -> coalesced columns
// ---------------------------------------------------------------------------
__global__ __launch_bounds__(256)
void ln_kernel(const float* __restrict__ in, const float* __restrict__ g,
               const float* __restrict__ bt, _Float16* __restrict__ out) {
  __shared__ float s_sum[256], s_sq[256];
  const int b = blockIdx.x;
  const int c = threadIdx.x & 127, half = threadIdx.x >> 7;
  const float* base = in + (size_t)b * kHW * kC;
  float sum = 0.f, sq = 0.f;
  for (int s = half * 392; s < (half + 1) * 392; ++s) {
    float v = base[(size_t)s * kC + c];
    sum += v; sq += v * v;
  }
  s_sum[threadIdx.x] = sum; s_sq[threadIdx.x] = sq;
  __syncthreads();
  if (half == 0) {
    sum += s_sum[threadIdx.x + 128];
    sq  += s_sq[threadIdx.x + 128];
    float m = sum * (1.0f / kHW);
    float v = sq * (1.0f / kHW) - m * m;
    s_sum[c] = m;
    s_sq[c]  = rsqrtf(v + 1e-5f);
  }
  __syncthreads();
  const float m = s_sum[c], inv = s_sq[c];
  for (int s = half * 392; s < (half + 1) * 392; ++s) {
    float v = (base[(size_t)s * kC + c] - m) * inv * g[s] + bt[s];
    out[((size_t)b * kHW + s) * kC + c] = (_Float16)v;
  }
}

// ---------------------------------------------------------------------------
// 2x2 stride-2 pad-1 depthwise conv on xn (f16 NHWC) for K and V in one pass.
// writes [B, 240, C] f16 with rows 225..239 zeroed.
// ---------------------------------------------------------------------------
__global__ __launch_bounds__(256)
void kv_dw_kernel(const _Float16* __restrict__ xn,
                  const float* __restrict__ kw, const float* __restrict__ kb,
                  const float* __restrict__ vw, const float* __restrict__ vb,
                  _Float16* __restrict__ kA, _Float16* __restrict__ vA) {
  int idx = blockIdx.x * 256 + threadIdx.x;   // over B*240*C (exact multiple)
  int c = idx & 127;
  int l = (idx >> 7) % kLp;
  int b = idx / (kLp * kC);
  if (l >= kL) { kA[idx] = (_Float16)0.f; vA[idx] = (_Float16)0.f; return; }
  int oy = l / kKVS, ox = l - oy * kKVS;
  float ka = kb[c], va = vb[c];
#pragma unroll
  for (int ky = 0; ky < 2; ++ky) {
    int iy = 2 * oy - 1 + ky;
    if (iy < 0 || iy >= kH) continue;
#pragma unroll
    for (int kx = 0; kx < 2; ++kx) {
      int ix = 2 * ox - 1 + kx;
      if (ix < 0 || ix >= kW) continue;
      float v = (float)xn[((size_t)b * kHW + iy * kW + ix) * kC + c];
      ka += kw[c * 4 + ky * 2 + kx] * v;
      va += vw[c * 4 + ky * 2 + kx] * v;
    }
  }
  kA[idx] = (_Float16)ka;
  vA[idx] = (_Float16)va;
}

// ---------------------------------------------------------------------------
// Generic WMMA GEMM: D[M,N] = epilogue(A[M,K] @ W[N,K]^T + bias[N]).
// A rows and W rows are both K-contiguous -> pure v16h loads, no LDS staging.
// 256 threads = 8 waves as 2(M)x4(N) 16x16 tiles; block tile 32x64.
// M,N are exact multiples of 32/64 for every call site.
// ---------------------------------------------------------------------------
enum { EP_F16 = 0, EP_F16_T = 1, EP_F32_RES = 2, EP_GELU_F16 = 3, EP_F32 = 4 };

template <int MODE>
__global__ __launch_bounds__(256)
void gemm_wmma(const _Float16* __restrict__ A, const _Float16* __restrict__ Wm,
               const float* __restrict__ bias, void* __restrict__ Dp,
               const float* __restrict__ R, int M, int N, int K, float scale) {
  const int wave = threadIdx.x >> 5, lane = threadIdx.x & 31;
  const int lr = lane & 15, lh = lane >> 4;
  const int mBase = blockIdx.x * 32 + (wave >> 2) * 16;
  const int nBase = blockIdx.y * 64 + (wave & 3) * 16;

  const _Float16* aP = A  + (size_t)(mBase + lr) * K + lh * 16;
  const _Float16* bP = Wm + (size_t)(nBase + lr) * K + lh * 16;
  v8f acc = {};
  for (int k = 0; k < K; k += 32) {
    v16h av = *(const v16h*)(aP + k);
    v16h bv = *(const v16h*)(bP + k);
    acc = WMMA_F16(av, bv, acc);
  }
  const int n = nBase + lr;
  const float bn = bias[n];
#pragma unroll
  for (int r = 0; r < 8; ++r) {
    const int m = mBase + r + lh * 8;
    float v = acc[r] + bn;
    if constexpr (MODE == EP_F16) {
      ((_Float16*)Dp)[(size_t)m * N + n] = (_Float16)(v * scale);
    } else if constexpr (MODE == EP_F16_T) {
      // V projection: store transposed as vT[b*128 + n][l] with l padded 256
      int bb = m / kLp, l = m - bb * kLp;
      ((_Float16*)Dp)[((size_t)bb * kC + n) * kLp2 + l] = (_Float16)v;
    } else if constexpr (MODE == EP_F32_RES) {
      ((float*)Dp)[(size_t)m * N + n] = v + R[(size_t)m * N + n];
    } else if constexpr (MODE == EP_GELU_F16) {
      ((_Float16*)Dp)[(size_t)m * N + n] = (_Float16)gelu_f(v);
    } else {
      ((float*)Dp)[(size_t)m * N + n] = v;
    }
  }
}

// ---------------------------------------------------------------------------
// Fused attention: one wave per (b, head, 16-row q tile).
//   scores  : 15 WMMAs (K=d=32), +attn_bias, mask cols>=225, f16 -> LDS
//   softmax : lanes 0..15 each own a row (f32 math)
//   P @ V   : 16 WMMAs over Kpad=256 against pre-transposed vT
// q is pre-scaled by 1/sqrt(d) in its projection epilogue.
// ---------------------------------------------------------------------------
__global__ __launch_bounds__(32)
void attn_kernel(const _Float16* __restrict__ q, const _Float16* __restrict__ kMat,
                 const _Float16* __restrict__ vT, const float* __restrict__ abias,
                 _Float16* __restrict__ o) {
  __shared__ __align__(32) _Float16 pBuf[16 * kLp2];
  const int blk = blockIdx.x;                // 49 * 4 * 64 blocks
  const int qt = blk % 49;
  const int h  = (blk / 49) & 3;
  const int b  = blk / (49 * 4);
  const int lane = threadIdx.x;
  const int lr = lane & 15, lh = lane >> 4;
  const int mBase = qt * 16;

  const v16h av = *(const v16h*)(q + ((size_t)b * kHW + mBase + lr) * kC + h * kD + lh * 16);

  for (int nt = 0; nt < 15; ++nt) {
    const v16h bv =
        *(const v16h*)(kMat + ((size_t)b * kLp + nt * 16 + lr) * kC + h * kD + lh * 16);
    v8f acc = {};
    acc = WMMA_F16(av, bv, acc);
    const int n = nt * 16 + lr;
#pragma unroll
    for (int r = 0; r < 8; ++r) {
      const int m = mBase + r + lh * 8;
      float val = (n < kL) ? acc[r] + abias[((size_t)h * kHW + m) * kL + n] : -30000.0f;
      pBuf[(r + lh * 8) * kLp2 + n] = (_Float16)val;
    }
  }
  __syncthreads();

  if (lane < 16) {
    float mx = -3.0e30f;
    for (int j = 0; j < kL; ++j) mx = fmaxf(mx, (float)pBuf[lane * kLp2 + j]);
    float sum = 0.f;
    for (int j = 0; j < kL; ++j) {
      float e = __expf((float)pBuf[lane * kLp2 + j] - mx);
      sum += e;
      pBuf[lane * kLp2 + j] = (_Float16)e;
    }
    const float inv = 1.0f / sum;
    for (int j = 0; j < kL; ++j)
      pBuf[lane * kLp2 + j] = (_Float16)((float)pBuf[lane * kLp2 + j] * inv);
    for (int j = kL; j < kLp2; ++j) pBuf[lane * kLp2 + j] = (_Float16)0.0f;
  }
  __syncthreads();

#pragma unroll
  for (int g2 = 0; g2 < 2; ++g2) {
    v8f acc = {};
    for (int kt = 0; kt < 8; ++kt) {
      v16h pa = *(const v16h*)(pBuf + lr * kLp2 + kt * 32 + lh * 16);
      v16h vb = *(const v16h*)(vT + ((size_t)b * kC + h * kD + g2 * 16 + lr) * kLp2 +
                               kt * 32 + lh * 16);
      acc = WMMA_F16(pa, vb, acc);
    }
#pragma unroll
    for (int r = 0; r < 8; ++r) {
      const int m = mBase + r + lh * 8;
      o[((size_t)b * kHW + m) * kC + h * kD + g2 * 16 + lr] = (_Float16)acc[r];
    }
  }
}

// ---------------------------------------------------------------------------
// training-mode BN stats over B*H*W per channel -> fused scale/shift
// ---------------------------------------------------------------------------
template <typename T>
__global__ __launch_bounds__(256)
void bn_stats(const T* __restrict__ in, const float* __restrict__ g,
              const float* __restrict__ bt, float* __restrict__ bs,
              float* __restrict__ bb, int Mtot, int N) {
  __shared__ float ss[256], sq[256];
  const int n = blockIdx.x;
  float s = 0.f, q = 0.f;
  for (int m = threadIdx.x; m < Mtot; m += 256) {
    float v = (float)in[(size_t)m * N + n];
    s += v; q += v * v;
  }
  ss[threadIdx.x] = s; sq[threadIdx.x] = q;
  __syncthreads();
  for (int o = 128; o > 0; o >>= 1) {
    if (threadIdx.x < o) { ss[threadIdx.x] += ss[threadIdx.x + o]; sq[threadIdx.x] += sq[threadIdx.x + o]; }
    __syncthreads();
  }
  if (threadIdx.x == 0) {
    float m = ss[0] / Mtot;
    float v = sq[0] / Mtot - m * m;
    float is = rsqrtf(v + 1e-5f);
    bs[n] = g[n] * is;
    bb[n] = bt[n] - m * g[n] * is;
  }
}

// ---------------------------------------------------------------------------
// FFN dwconv: h2g = gelu(dwconv3x3(BN1(h1g)) + BN1(h1g))   (channel-fast, E=512)
// ---------------------------------------------------------------------------
__global__ __launch_bounds__(256)
void ffn_dw_kernel(const _Float16* __restrict__ h1g, const float* __restrict__ bs1,
                   const float* __restrict__ bb1, const float* __restrict__ dww,
                   const float* __restrict__ dwb, _Float16* __restrict__ h2g) {
  size_t idx = (size_t)blockIdx.x * 256 + threadIdx.x;   // B*HW*E, exact multiple
  const int e = (int)(idx & 511);
  const int s = (int)((idx >> 9) % kHW);
  const int b = (int)(idx / ((size_t)kHW * kE));
  const int yy = s / kW, xx = s - yy * kW;
  const float s1 = bs1[e], b1 = bb1[e];
  const float center = (float)h1g[idx] * s1 + b1;
  float acc = dwb[e] + center;
#pragma unroll
  for (int ky = 0; ky < 3; ++ky) {
    int iy = yy + ky - 1;
    if (iy < 0 || iy >= kH) continue;
#pragma unroll
    for (int kx = 0; kx < 3; ++kx) {
      int ix = xx + kx - 1;
      if (ix < 0 || ix >= kW) continue;
      float v = (float)h1g[((size_t)b * kHW + iy * kW + ix) * kE + e] * s1 + b1;
      acc += dww[e * 9 + ky * 3 + kx] * v;
    }
  }
  h2g[idx] = (_Float16)gelu_f(acc);
}

__global__ __launch_bounds__(256)
void bn_apply_f16(_Float16* __restrict__ buf, const float* __restrict__ bs,
                  const float* __restrict__ bb, int N, size_t total) {
  size_t i = (size_t)blockIdx.x * 256 + threadIdx.x;
  if (i >= total) return;
  int n = (int)(i % N);
  buf[i] = (_Float16)((float)buf[i] * bs[n] + bb[n]);
}

// ---------------------------------------------------------------------------
// final: out(NCHW) = BN3(h3pre) + x_mhsa   (one block per (b,c) plane)
// ---------------------------------------------------------------------------
__global__ __launch_bounds__(256)
void final_kernel(const float* __restrict__ h3, const float* __restrict__ bs,
                  const float* __restrict__ bb, const float* __restrict__ xm,
                  float* __restrict__ out) {
  const int b = blockIdx.x >> 7, c = blockIdx.x & 127;
  const float s3 = bs[c], b3 = bb[c];
  for (int s = threadIdx.x; s < kHW; s += 256) {
    size_t ix = ((size_t)b * kHW + s) * kC + c;
    out[((size_t)b * kC + c) * kHW + s] = h3[ix] * s3 + b3 + xm[ix];
  }
}

// ---------------------------------------------------------------------------
// host launch
// ---------------------------------------------------------------------------
extern "C" void kernel_launch(void* const* d_in, const int* in_sizes, int n_in,
                              void* d_out, int out_size, void* d_ws, size_t ws_size,
                              hipStream_t stream) {
  (void)in_sizes; (void)n_in; (void)out_size; (void)ws_size;

  const float* x      = (const float*)d_in[0];
  const float* lpu_w  = (const float*)d_in[1];
  const float* lpu_b  = (const float*)d_in[2];
  const float* ln1_g  = (const float*)d_in[3];
  const float* ln1_b  = (const float*)d_in[4];
  const float* wq     = (const float*)d_in[5];
  const float* bq     = (const float*)d_in[6];
  const float* wk     = (const float*)d_in[7];
  const float* bk     = (const float*)d_in[8];
  const float* wv     = (const float*)d_in[9];
  const float* bv     = (const float*)d_in[10];
  const float* wo     = (const float*)d_in[11];
  const float* bo     = (const float*)d_in[12];
  const float* kdw_w  = (const float*)d_in[13];
  const float* kdw_b  = (const float*)d_in[14];
  const float* vdw_w  = (const float*)d_in[15];
  const float* vdw_b  = (const float*)d_in[16];
  const float* abias  = (const float*)d_in[17];
  const float* ln2_g  = (const float*)d_in[18];
  const float* ln2_b  = (const float*)d_in[19];
  const float* c1_w   = (const float*)d_in[20];
  const float* c1_b   = (const float*)d_in[21];
  const float* bn1_g  = (const float*)d_in[22];
  const float* bn1_b  = (const float*)d_in[23];
  const float* dw_w   = (const float*)d_in[24];
  const float* dw_b   = (const float*)d_in[25];
  const float* bnr_g  = (const float*)d_in[26];
  const float* bnr_b  = (const float*)d_in[27];
  const float* c2_w   = (const float*)d_in[28];
  const float* c2_b   = (const float*)d_in[29];
  const float* bn2_g  = (const float*)d_in[30];
  const float* bn2_b  = (const float*)d_in[31];

  char* ws = (char*)d_ws;
  auto alloc = [&](size_t bytes) -> char* {
    char* p = ws;
    ws += (bytes + 255) & ~(size_t)255;
    return p;
  };

  float*     x_lpu  = (float*)    alloc((size_t)kM * kC * 4);
  float*     x_mhsa = (float*)    alloc((size_t)kM * kC * 4);
  float*     h3pre  = (float*)    alloc((size_t)kM * kC * 4);
  _Float16*  xn     = (_Float16*) alloc((size_t)kM * kC * 2);
  _Float16*  qF     = (_Float16*) alloc((size_t)kM * kC * 2);
  _Float16*  yF     = (_Float16*) alloc((size_t)kM * kC * 2);
  _Float16*  oF     = (_Float16*) alloc((size_t)kM * kC * 2);
  _Float16*  kA     = (_Float16*) alloc((size_t)kB * kLp * kC * 2);
  _Float16*  vA     = (_Float16*) alloc((size_t)kB * kLp * kC * 2);
  _Float16*  kMat   = (_Float16*) alloc((size_t)kB * kLp * kC * 2);
  _Float16*  vT     = (_Float16*) alloc((size_t)kB * kC * kLp2 * 2);
  _Float16*  h1g    = (_Float16*) alloc((size_t)kM * kE * 2);
  _Float16*  h2g    = (_Float16*) alloc((size_t)kM * kE * 2);
  _Float16*  wq16   = (_Float16*) alloc(kC * kC * 2);
  _Float16*  wk16   = (_Float16*) alloc(kC * kC * 2);
  _Float16*  wv16   = (_Float16*) alloc(kC * kC * 2);
  _Float16*  wo16   = (_Float16*) alloc(kC * kC * 2);
  _Float16*  c1w16  = (_Float16*) alloc(kE * kC * 2);
  _Float16*  c2w16  = (_Float16*) alloc(kC * kE * 2);
  float*     bnS1   = (float*)    alloc(kE * 4);
  float*     bnB1   = (float*)    alloc(kE * 4);
  float*     bnS2   = (float*)    alloc(kE * 4);
  float*     bnB2   = (float*)    alloc(kE * 4);
  float*     bnS3   = (float*)    alloc(kC * 4);
  float*     bnB3   = (float*)    alloc(kC * 4);

  // weights -> f16
  cvt_f32_to_f16<<<(kC * kC + 255) / 256, 256, 0, stream>>>(wq, wq16, kC * kC);
  cvt_f32_to_f16<<<(kC * kC + 255) / 256, 256, 0, stream>>>(wk, wk16, kC * kC);
  cvt_f32_to_f16<<<(kC * kC + 255) / 256, 256, 0, stream>>>(wv, wv16, kC * kC);
  cvt_f32_to_f16<<<(kC * kC + 255) / 256, 256, 0, stream>>>(wo, wo16, kC * kC);
  cvt_f32_to_f16<<<(kE * kC + 255) / 256, 256, 0, stream>>>(c1_w, c1w16, kE * kC);
  cvt_f32_to_f16<<<(kC * kE + 255) / 256, 256, 0, stream>>>(c2_w, c2w16, kC * kE);

  // LPU + LN1
  lpu_kernel<<<kB * kC, 256, 0, stream>>>(x, lpu_w, lpu_b, x_lpu);
  ln_kernel<<<kB, 256, 0, stream>>>(x_lpu, ln1_g, ln1_b, xn);

  // Q projection (scale 1/sqrt(d) folded into epilogue)
  gemm_wmma<EP_F16><<<dim3(kM / 32, kC / 64), 256, 0, stream>>>(
      xn, wq16, bq, qF, nullptr, kM, kC, kC, 0.17677669529663687f);

  // reduced K/V tokens
  kv_dw_kernel<<<(kB * kLp * kC) / 256, 256, 0, stream>>>(xn, kdw_w, kdw_b, vdw_w, vdw_b,
                                                          kA, vA);
  gemm_wmma<EP_F16><<<dim3((kB * kLp) / 32, kC / 64), 256, 0, stream>>>(
      kA, wk16, bk, kMat, nullptr, kB * kLp, kC, kC, 1.0f);
  fill_zero_f16<<<(kB * kC * kLp2) / 256, 256, 0, stream>>>(vT, kB * kC * kLp2);
  gemm_wmma<EP_F16_T><<<dim3((kB * kLp) / 32, kC / 64), 256, 0, stream>>>(
      vA, wv16, bv, vT, nullptr, kB * kLp, kC, kC, 1.0f);

  // fused attention (softmax in LDS, WMMA score + PV)
  attn_kernel<<<kB * kHeads * (kHW / 16), 32, 0, stream>>>(qF, kMat, vT, abias, oF);

  // output projection + residual with x_lpu
  gemm_wmma<EP_F32_RES><<<dim3(kM / 32, kC / 64), 256, 0, stream>>>(
      oF, wo16, bo, x_mhsa, x_lpu, kM, kC, kC, 1.0f);

  // LN2
  ln_kernel<<<kB, 256, 0, stream>>>(x_mhsa, ln2_g, ln2_b, yF);

  // FFN: expand + GELU
  gemm_wmma<EP_GELU_F16><<<dim3(kM / 32, kE / 64), 256, 0, stream>>>(
      yF, c1w16, c1_b, h1g, nullptr, kM, kE, kC, 1.0f);
  bn_stats<_Float16><<<kE, 256, 0, stream>>>(h1g, bn1_g, bn1_b, bnS1, bnB1, kM, kE);

  // dwconv + residual + GELU
  ffn_dw_kernel<<<(int)(((size_t)kM * kE) / 256), 256, 0, stream>>>(h1g, bnS1, bnB1, dw_w,
                                                                    dw_b, h2g);
  bn_stats<_Float16><<<kE, 256, 0, stream>>>(h2g, bnr_g, bnr_b, bnS2, bnB2, kM, kE);
  bn_apply_f16<<<(int)(((size_t)kM * kE) / 256), 256, 0, stream>>>(h2g, bnS2, bnB2, kE,
                                                                   (size_t)kM * kE);

  // FFN: project down
  gemm_wmma<EP_F32><<<dim3(kM / 32, kC / 64), 256, 0, stream>>>(
      h2g, c2w16, c2_b, h3pre, nullptr, kM, kC, kE, 1.0f);
  bn_stats<float><<<kC, 256, 0, stream>>>(h3pre, bn2_g, bn2_b, bnS3, bnB3, kM, kC);

  // BN3 + residual, NHWC -> NCHW
  final_kernel<<<kB * kC, 256, 0, stream>>>(h3pre, bnS3, bnB3, x_mhsa, (float*)d_out);
}
